// AggregationAttentionLayer_37873021616623
// MI455X (gfx1250) — compile-verified, matrix operand order checked by reading, not run on hardware
//
#include <hip/hip_runtime.h>
#include <hip/hip_bf16.h>

// ---------------------------------------------------------------------------
// AggregationAttentionLayer fused kernel for MI455X (gfx1250, wave32, WMMA).
//
// Dominant cost: K/V projections (targets[78400,512] @ W[512,512] x2, 82 GFLOP)
// -> done with v_wmma_f32_16x16x32_bf16 (8x the K-depth of the fp32 WMMA).
// targets is read from HBM exactly once (160 MB), converted to bf16 in LDS.
// ---------------------------------------------------------------------------

#define BATCH   8
#define NQ      196
#define NK      50
#define DM      512          // d_model == H*D_INTER
#define HEADS   8
#define DI      64
#define NBQ     (BATCH * NQ) // 1568 workgroups

#define MT      64           // padded M (nk rows padded 50 -> 64)
#define TSTR    528          // LDS row stride in bf16 elems: 1056 B (16B aligned, 8-bank skew)

typedef __attribute__((ext_vector_type(16))) __bf16 v16bf;
typedef __attribute__((ext_vector_type(8)))  float  v8f;

struct B32x8 { uint4 lo, hi; };   // 32 bytes == one v16bf fragment

__device__ __forceinline__ unsigned short f2bf(float f) {
    unsigned u = __builtin_bit_cast(unsigned, f);
    u += 0x7FFFu + ((u >> 16) & 1u);          // round-to-nearest-even
    return (unsigned short)(u >> 16);
}
__device__ __forceinline__ float bf2f(unsigned short h) {
    return __builtin_bit_cast(float, ((unsigned)h) << 16);
}

// ---------------------------------------------------------------------------
// Pack Wk/Wv (fp32, 512x512, row=K dim, col=N dim) into the per-lane bf16
// B-fragment layout of V_WMMA_F32_16X16X32_BF16. Fragment f = ktile*32+ntile,
// lane l holds column N = ntile*16 + (l&15); dword v holds the K pair
//   K = ktile*32 + (v<4 ? 0 : 16) + ((l&16)?8:0) + 2*(v&3) + {0,1}.
// Packed address: dword index ((f*32 + lane)*8 + v)  -> 32 contiguous bytes
// per lane per fragment => two global_load_b128 in the GEMM inner loop.
// ---------------------------------------------------------------------------
__global__ __launch_bounds__(256) void pack_weights_kernel(
        const float* __restrict__ Wk, const float* __restrict__ Wv,
        unsigned* __restrict__ dst)
{
    int gid = blockIdx.x * 256 + threadIdx.x;      // one packed dword each
    const int PER_MAT = 16 * 32 * 32 * 8;          // 131072 dwords per matrix
    if (gid >= 2 * PER_MAT) return;
    int m    = gid / PER_MAT;
    int r    = gid % PER_MAT;
    int f    = r >> 8;                              // fragment index
    int lane = (r >> 3) & 31;
    int v    = r & 7;
    int kt   = f >> 5, nt = f & 31;
    int kk   = (v & 4 ? 16 : 0) + ((lane & 16) ? 8 : 0) + 2 * (v & 3);
    int row  = kt * 32 + kk;
    int col  = nt * 16 + (lane & 15);
    const float* W = m ? Wv : Wk;
    unsigned short lo = f2bf(W[(size_t)row       * DM + col]);
    unsigned short hi = f2bf(W[(size_t)(row + 1) * DM + col]);
    dst[gid] = (unsigned)lo | ((unsigned)hi << 16);
}

// ---------------------------------------------------------------------------
// Fused per-(b,q) kernel: 256 threads = 8 waves.
//   phase 1: targets slice (50x512 fp32) -> bf16 LDS (padded to 64 rows)
//   phase 1b: q = queries_row @ Wq + bq (scalar, coalesced weight reads)
//   phase 2: K|V = T @ {Wk,Wv} + b  via WMMA (wave -> {matrix, m-tile})
//   phase 3: scores -> softmax(nk) -> att @ V -> out @ Wo + bo
// ---------------------------------------------------------------------------
__global__ __launch_bounds__(256) void aggattn_fused_kernel(
        const float* __restrict__ queries, const float* __restrict__ targets,
        const float* __restrict__ Wq,      const float* __restrict__ bq,
        const float* __restrict__ bk,      const float* __restrict__ bv,
        const float* __restrict__ Wo,      const float* __restrict__ bo,
        const unsigned* __restrict__ WkP,  const unsigned* __restrict__ WvP,
        float* __restrict__ out)
{
    __shared__ __align__(16) unsigned short Tl[MT * TSTR];   // 67584 B
    __shared__ __align__(16) unsigned short Kl[MT * TSTR];   // 67584 B
    __shared__ __align__(16) unsigned short Vl[MT * TSTR];   // 67584 B
    __shared__ __align__(16) float qrow[DM];
    __shared__ __align__(16) float qp[DM];
    __shared__ __align__(16) float att[HEADS * 64];
    __shared__ __align__(16) float outv[DM];

    const int bqi = blockIdx.x;
    const int tid = threadIdx.x;

    // ---- phase 1: stage targets tile as bf16 (zero the 14 pad rows) -------
    const float* Tg = targets + (size_t)bqi * NK * DM;
    for (int i = tid; i < MT * DM; i += 256) {
        int r = i >> 9, c = i & (DM - 1);
        Tl[r * TSTR + c] = (r < NK) ? f2bf(Tg[(size_t)r * DM + c]) : (unsigned short)0;
    }
    const float* qg = queries + (size_t)bqi * DM;
    for (int i = tid; i < DM; i += 256) qrow[i] = qg[i];
    __syncthreads();

    // ---- phase 1b: q projection (0.8 GFLOP total; coalesced over j) -------
    for (int j = tid; j < DM; j += 256) {
        float s = bq[j];
        #pragma unroll 8
        for (int i = 0; i < DM; ++i) s += qrow[i] * Wq[(size_t)i * DM + j];
        qp[j] = s;
    }

    // ---- phase 2: K and V projections via WMMA ----------------------------
    const int wave  = tid >> 5;
    const int lane  = tid & 31;
    const int mtile = wave & 3;        // 4 m-tiles of 16 rows
    const int isV   = wave >> 2;       // waves 0-3 -> K, waves 4-7 -> V
    const unsigned*       Wp   = isV ? WvP : WkP;
    const float*          bias = isV ? bv  : bk;
    unsigned short*       Ol   = isV ? Vl  : Kl;

    // A fragments for this wave's 16 rows, all 16 k-steps (held in VGPRs).
    const int arow = mtile * 16 + (lane & 15);
    const unsigned short* abase = Tl + (size_t)arow * TSTR + ((lane & 16) ? 8 : 0);
    v16bf afrag[16];
    #pragma unroll
    for (int k = 0; k < 16; ++k) {
        B32x8 t;
        t.lo = *(const uint4*)(abase + k * 32);        // K = k*32 + [0..7]+kb
        t.hi = *(const uint4*)(abase + k * 32 + 16);   // K = k*32 + 16 + kb
        afrag[k] = __builtin_bit_cast(v16bf, t);
    }

    const float inv_scale = 1.0f / 8.000001f;          // 1/(sqrt(64)+1e-6)

    for (int nt = 0; nt < 32; ++nt) {                  // 32 n-tiles of 16 cols
        v8f acc = {};
        const unsigned* bp = Wp + (((size_t)nt * 32 + lane) << 3);
        #pragma unroll
        for (int k = 0; k < 16; ++k) {                 // k-tile stride: 32*32*8 dwords
            const unsigned* p = bp + ((size_t)k << 13);
            B32x8 t;
            t.lo = *(const uint4*)(p);
            t.hi = *(const uint4*)(p + 4);
            v16bf bfrag = __builtin_bit_cast(v16bf, t);
            acc = __builtin_amdgcn_wmma_f32_16x16x32_bf16(
                      false, afrag[k], false, bfrag, (short)0, acc, false, false);
        }
        const int col   = nt * 16 + (lane & 15);
        const float bb  = bias[col];
        const int rbase = mtile * 16 + ((lane & 16) ? 8 : 0);
        #pragma unroll
        for (int r = 0; r < 8; ++r)
            Ol[(size_t)(rbase + r) * TSTR + col] = f2bf(acc[r] + bb);
    }
    __syncthreads();

    // ---- phase 3a: attention scores  att[h,k] = <q_h, K_k,h>/scale --------
    for (int i = tid; i < HEADS * NK; i += 256) {
        int h = i / NK, kk = i % NK;
        const unsigned short* kp = Kl + (size_t)kk * TSTR + h * DI;
        const float* qh = qp + h * DI;
        float s = 0.f;
        #pragma unroll 8
        for (int d = 0; d < DI; ++d) s += qh[d] * bf2f(kp[d]);
        att[h * 64 + kk] = s * inv_scale;
    }
    __syncthreads();

    // ---- phase 3b: softmax over nk (one thread per head; nk=50, trivial) --
    if (tid < HEADS) {
        float* a = att + tid * 64;
        float mx = a[0];
        for (int k = 1; k < NK; ++k) mx = fmaxf(mx, a[k]);
        float sum = 0.f;
        for (int k = 0; k < NK; ++k) { float e = __expf(a[k] - mx); a[k] = e; sum += e; }
        float inv = 1.0f / sum;
        for (int k = 0; k < NK; ++k) a[k] *= inv;
    }
    __syncthreads();

    // ---- phase 3c: out_h = sum_k att[h,k] * V[k,h,:] ----------------------
    for (int j = tid; j < DM; j += 256) {
        int h = j >> 6;
        const float* a = att + h * 64;
        float s = 0.f;
        #pragma unroll 5
        for (int kk = 0; kk < NK; ++kk) s += a[kk] * bf2f(Vl[(size_t)kk * TSTR + j]);
        outv[j] = s;
    }
    __syncthreads();

    // ---- phase 3d: final projection  y = out @ Wo + bo --------------------
    float* og = out + (size_t)bqi * DM;
    for (int j = tid; j < DM; j += 256) {
        float s = bo[j];
        #pragma unroll 8
        for (int c = 0; c < DM; ++c) s += outv[c] * Wo[(size_t)c * DM + j];
        og[j] = s;
    }
}

extern "C" void kernel_launch(void* const* d_in, const int* in_sizes, int n_in,
                              void* d_out, int out_size, void* d_ws, size_t ws_size,
                              hipStream_t stream) {
    const float* queries = (const float*)d_in[0];
    const float* targets = (const float*)d_in[1];
    const float* Wq      = (const float*)d_in[2];
    const float* bq      = (const float*)d_in[3];
    const float* Wk      = (const float*)d_in[4];
    const float* bk      = (const float*)d_in[5];
    const float* Wv      = (const float*)d_in[6];
    const float* bv      = (const float*)d_in[7];
    const float* Wo      = (const float*)d_in[8];
    const float* bo      = (const float*)d_in[9];
    float*       out     = (float*)d_out;

    unsigned* wsPacked = (unsigned*)d_ws;              // 1 MB: Wk then Wv, bf16 frags
    const int PER_MAT  = 16 * 32 * 32 * 8;             // 131072 dwords / matrix
    const int packN    = 2 * PER_MAT;

    pack_weights_kernel<<<(packN + 255) / 256, 256, 0, stream>>>(Wk, Wv, wsPacked);
    aggattn_fused_kernel<<<NBQ, 256, 0, stream>>>(
        queries, targets, Wq, bq, bk, bv, Wo, bo,
        wsPacked, wsPacked + PER_MAT, out);
}
